// ARMA_85976655332070
// MI455X (gfx1250) — compile-verified, matrix operand order checked by reading, not run on hardware
//
#include <hip/hip_runtime.h>
#include <hip/hip_bf16.h>

// ---------------------------------------------------------------------------
// ARMA-GCN forward for MI455X (gfx1250, wave32).
//  - All activations kept in bf16 in workspace (halves L2/HBM traffic of the
//    dominant streams: GEMM A reads and SpMM row gathers).
//  - Dense GEMMs: v_wmma_f32_16x16x32_bf16, f32 accumulate, bf16 outputs.
//    Weights pre-transposed/converted once per launch; B chunk staged once
//    per 128-K block; A tiles double-buffered through LDS.
//  - SpMM: wave-per-edge bf16 row gather (8B/lane) + f32 atomics into the
//    L2-resident accumulator (51.2MB << 192MB L2).
// ---------------------------------------------------------------------------

#define N_F   512
#define N_H   128
#define N_C   40

typedef __attribute__((ext_vector_type(16))) __bf16 v16bf;
typedef __attribute__((ext_vector_type(8)))  float  v8f;

__device__ __forceinline__ unsigned short f2bf_rne(float f) {
  union { float f; unsigned int u; } c; c.f = f;
  unsigned int u = c.u;
  u += 0x7fffu + ((u >> 16) & 1u);          // round-to-nearest-even
  return (unsigned short)(u >> 16);
}
__device__ __forceinline__ float bf2f(unsigned short s) {
  union { unsigned int u; float f; } c; c.u = ((unsigned int)s) << 16; return c.f;
}
__device__ __forceinline__ __bf16 u2bf(unsigned short s) {
  union { unsigned short s; __bf16 b; } c; c.s = s; return c.b;
}

// ---------------------------------------------------------------------------
// GEMM: out[M,NOUT] = epilogue(A[M,KIN]_bf16 @ W[KIN,NOUT])
//   WT is the pre-transposed bf16 weight, layout [NPAD][KIN].
//   epilogue: (+ add[M,NOUT] f32) (+ bias[NOUT] f32) (relu), out bf16 or f32.
// Block: 256 threads (8 waves), tile 128 rows x NOUT cols.
// K loop: 128-wide B chunks staged once; 32-wide A tiles double-buffered.
// ---------------------------------------------------------------------------
template<int KIN, int NOUT, bool RELU, bool HAS_ADD, bool HAS_BIAS, bool OUT_F32>
__global__ __launch_bounds__(256)
void gemm_bf16_wmma(const unsigned short* __restrict__ A,
                    const unsigned short* __restrict__ WT,
                    const float* __restrict__ bias,
                    const float* __restrict__ add,
                    void* __restrict__ outv, int M) {
  constexpr int NT    = (NOUT + 15) / 16;
  constexpr int NPAD  = NT * 16;
  constexpr int KBLK  = 128;                 // B chunk width (KIN % 128 == 0)
  constexpr int NKB   = KIN / KBLK;
  constexpr int NKTOT = KIN / 32;
  constexpr int SA_S  = 40;                  // 32 + 8 pad: 80B rows, 20-bank stride
  constexpr int SB_S  = KBLK + 8;            // 272B rows, stride = 4 banks mod 64

  __shared__ unsigned short sA[2][128 * SA_S];
  __shared__ unsigned short sB[NPAD * SB_S];

  const int tid  = threadIdx.x;
  const int wave = tid >> 5;
  const int lane = tid & 31;
  const int half = lane >> 4;                // ISA 7.12.2: lanes 0-15 vs 16-31
  const int ml   = lane & 15;
  const int row0 = blockIdx.x * 128;

  // A tile loader: 128 rows x 32 bf16 = 8KB; 2 x uint4 (16B) per thread.
  const int a_row = tid >> 1;                // unused; load_a: j=0 -> rows 0..63, j=1 -> rows 64..127 (4 chunks/row)
  uint4 areg[2];
  auto load_a = [&](int k0) {
#pragma unroll
    for (int j = 0; j < 2; ++j) {
      int chunk = tid + j * 256;             // 512 chunks of 8 shorts
      int row = chunk >> 2;
      int co  = (chunk & 3) * 8;
      int gr  = row0 + row;
      if (gr < M) areg[j] = *(const uint4*)&A[(size_t)gr * KIN + k0 + co];
      else        areg[j] = make_uint4(0u, 0u, 0u, 0u);
    }
  };
  auto store_a = [&](int buf) {
#pragma unroll
    for (int j = 0; j < 2; ++j) {
      int chunk = tid + j * 256;
      int row = chunk >> 2;
      int co  = (chunk & 3) * 8;
      *(uint4*)&sA[buf][row * SA_S + co] = areg[j];
    }
  };
  (void)a_row;

  v8f acc[NT];
#pragma unroll
  for (int nt = 0; nt < NT; ++nt)
#pragma unroll
    for (int j = 0; j < 8; ++j) acc[nt][j] = 0.f;

  load_a(0);
  store_a(0);

  for (int kb = 0; kb < NKB; ++kb) {
    // stage B chunk [NPAD x 128] once per 128-K block (uint4 copies, remapped
    // into the padded stride). Prior barrier guarantees old reads finished.
#pragma unroll
    for (int g = tid; g < NPAD * (KBLK / 8); g += 256) {
      int col = g / (KBLK / 8);
      int ko  = (g % (KBLK / 8)) * 8;
      *(uint4*)&sB[col * SB_S + ko] =
          *(const uint4*)&WT[(size_t)col * KIN + kb * KBLK + ko];
    }
    __syncthreads();

#pragma unroll
    for (int ki = 0; ki < KBLK / 32; ++ki) {
      const int ks  = kb * (KBLK / 32) + ki;
      const int buf = ks & 1;

      if (ks + 1 < NKTOT) load_a((ks + 1) * 32);   // prefetch next tile to regs
      if (ks + 2 < NKTOT)                          // and hint L2 two steps ahead
        __builtin_prefetch(&A[(size_t)(row0 + (tid >> 1)) * KIN + (ks + 2) * 32], 0, 1);

      // A fragment: lane<16 -> row ml, K{0..7,16..23}; lane>=16 -> K{8..15,24..31}
      v16bf afrag;
      const unsigned short* aptr = sA[buf] + (wave * 16 + ml) * SA_S;
#pragma unroll
      for (int j = 0; j < 8; ++j) {
        int i  = 2 * j;
        int kk = i + 8 * half + ((i & 8) ? 8 : 0);   // even -> 4B-aligned pair
        unsigned int w2 = *(const unsigned int*)(aptr + kk);
        afrag[i]     = u2bf((unsigned short)(w2 & 0xffffu));
        afrag[i + 1] = u2bf((unsigned short)(w2 >> 16));
      }

#pragma unroll
      for (int nt = 0; nt < NT; ++nt) {
        v16bf bfrag;
        const unsigned short* bptr = sB + (nt * 16 + ml) * SB_S + ki * 32;
#pragma unroll
        for (int j = 0; j < 8; ++j) {
          int i  = 2 * j;
          int kk = i + 8 * half + ((i & 8) ? 8 : 0);
          unsigned int w2 = *(const unsigned int*)(bptr + kk);
          bfrag[i]     = u2bf((unsigned short)(w2 & 0xffffu));
          bfrag[i + 1] = u2bf((unsigned short)(w2 >> 16));
        }
        acc[nt] = __builtin_amdgcn_wmma_f32_16x16x32_bf16(
            false, afrag, false, bfrag, (short)0, acc[nt], false, false);
      }

      if (ks + 1 < NKTOT) {
        store_a(buf ^ 1);        // other buffer: safe while peers still read buf
        __syncthreads();         // single barrier per K-step
      }
    }
  }

  // epilogue (C/D layout: VGPR r -> M=r (lanes 0-15) / M=r+8 (lanes 16-31))
#pragma unroll
  for (int nt = 0; nt < NT; ++nt) {
#pragma unroll
    for (int r = 0; r < 8; ++r) {
      int gr  = row0 + wave * 16 + half * 8 + r;
      int col = nt * 16 + ml;
      if (gr < M && col < NOUT) {
        float v = acc[nt][r];
        if (HAS_ADD)  v += add[(size_t)gr * NOUT + col];
        if (HAS_BIAS) v += bias[col];
        if (RELU)     v = fmaxf(v, 0.f);
        if constexpr (OUT_F32)
          ((float*)outv)[(size_t)gr * NOUT + col] = v;
        else
          ((unsigned short*)outv)[(size_t)gr * NOUT + col] = f2bf_rne(v);
      }
    }
  }
}

// ---------------------------------------------------------------------------
// Preprocessing kernels
// ---------------------------------------------------------------------------
__global__ void f32_to_bf16_kernel(const float* __restrict__ in,
                                   unsigned short* __restrict__ out, long long n) {
  long long i = (long long)blockIdx.x * blockDim.x + threadIdx.x;
  if (i < n) out[i] = f2bf_rne(in[i]);
}

// W [KIN][NOUT] f32 -> WT [NPAD][KIN] bf16, zero-padded columns.
__global__ void transpose_w_kernel(const float* __restrict__ W,
                                   unsigned short* __restrict__ WT,
                                   int KIN, int NOUT, int NPAD) {
  int idx = blockIdx.x * blockDim.x + threadIdx.x;
  if (idx >= NPAD * KIN) return;
  int col = idx / KIN, k = idx % KIN;
  float v = (col < NOUT) ? W[(size_t)k * NOUT + col] : 0.f;
  WT[(size_t)col * KIN + k] = f2bf_rne(v);
}

// ---------------------------------------------------------------------------
// Graph kernels
// ---------------------------------------------------------------------------
__global__ void deg_kernel(const int* __restrict__ dst,
                           const float* __restrict__ ew,
                           float* __restrict__ deg, int E) {
  int e = blockIdx.x * blockDim.x + threadIdx.x;
  if (e < E) atomicAdd(deg + dst[e], ew[e]);
}

__global__ void dis_kernel(const float* __restrict__ deg,
                           float* __restrict__ dis, int n) {
  int i = blockIdx.x * blockDim.x + threadIdx.x;
  if (i < n) {
    float d = deg[i];
    dis[i] = (d > 0.f) ? rsqrtf(fmaxf(d, 1e-30f)) : 0.f;
  }
}

__global__ void norm_kernel(const int* __restrict__ src,
                            const int* __restrict__ dst,
                            const float* __restrict__ ew,
                            const float* __restrict__ dis,
                            float* __restrict__ nrm, int E) {
  int e = blockIdx.x * blockDim.x + threadIdx.x;
  if (e < E) nrm[e] = dis[src[e]] * ew[e] * dis[dst[e]];
}

// agg[dst] += norm * feat[src] (feat bf16). One wave per edge: 32 lanes x
// 4 bf16 (8B) = the full 128-feature row, fully coalesced; f32 atomics.
__global__ void spmm_kernel(const unsigned short* __restrict__ feat,
                            const int* __restrict__ src,
                            const int* __restrict__ dst,
                            const float* __restrict__ nrm,
                            float* __restrict__ agg, int E) {
  long long gid = (long long)blockIdx.x * blockDim.x + threadIdx.x;
  long long e = gid >> 5;
  if (e >= E) return;
  int lane = (int)(gid & 31);
  int s = src[e], d = dst[e];
  float w = nrm[e];
  ushort4 v = *(const ushort4*)(feat + (size_t)s * N_H + lane * 4);
  float* op = agg + (size_t)d * N_H + lane * 4;
  atomicAdd(op + 0, bf2f(v.x) * w);
  atomicAdd(op + 1, bf2f(v.y) * w);
  atomicAdd(op + 2, bf2f(v.z) * w);
  atomicAdd(op + 3, bf2f(v.w) * w);
}

__global__ void mean_relu_kernel(const unsigned short* __restrict__ a,
                                 const unsigned short* __restrict__ b,
                                 unsigned short* __restrict__ o, long long n) {
  long long i = (long long)blockIdx.x * blockDim.x + threadIdx.x;
  if (i < n) o[i] = f2bf_rne(fmaxf(0.5f * (bf2f(a[i]) + bf2f(b[i])), 0.f));
}

// ---------------------------------------------------------------------------
// Host orchestration
// ---------------------------------------------------------------------------
template<int KIN, int NOUT, bool RELU, bool HAS_ADD, bool HAS_BIAS, bool OUT_F32>
static void launch_gemm(const unsigned short* A, const unsigned short* WT,
                        const float* bias, const float* add, void* out, int M,
                        hipStream_t s) {
  int grid = (M + 127) / 128;
  hipLaunchKernelGGL((gemm_bf16_wmma<KIN, NOUT, RELU, HAS_ADD, HAS_BIAS, OUT_F32>),
                     dim3(grid), dim3(256), 0, s, A, WT, bias, add, out, M);
}

extern "C" void kernel_launch(void* const* d_in, const int* in_sizes, int n_in,
                              void* d_out, int out_size, void* d_ws, size_t ws_size,
                              hipStream_t stream) {
  const float* x      = (const float*)d_in[0];   // [N, 512]
  const int*   eidx   = (const int*)  d_in[1];   // [2, E]
  const float* ew     = (const float*)d_in[2];   // [E]
  const float* Win    = (const float*)d_in[3];   // [512, 128]
  const float* b_in   = (const float*)d_in[4];   // [128]
  const float* init_w = (const float*)d_in[5];   // [L, K, 128, 128]
  const float* w_deep = (const float*)d_in[6];   // [L, 1, K, 128, 128]
  const float* root_w = (const float*)d_in[7];   // [L, T, K, 128, 128]
  const float* bias   = (const float*)d_in[8];   // [L, T, K, 1, 128]
  const float* Wout   = (const float*)d_in[9];   // [128, 40]
  const float* b_out  = (const float*)d_in[10];  // [40]
  float* out = (float*)d_out;                    // [N, 40]

  const int N = in_sizes[0] / N_F;
  const int E = in_sizes[2];
  const int* src = eidx;        // edge_index[0] = row
  const int* dst = eidx + E;    // edge_index[1] = col
  const long long NH = (long long)N * N_H;
  const int L = 2, K = 2, T = 2, HH = N_H * N_H;
  const int NPAD_C = 48;        // padded class count for output GEMM

  // ---- workspace layout (256B-aligned carve-out) ----
  char* cur = (char*)d_ws;
  auto carve = [&](size_t bytes) {
    void* r = (void*)cur;
    cur += (bytes + 255) & ~(size_t)255;
    return r;
  };
  float* deg = (float*)carve((size_t)N * 4);
  float* dis = (float*)carve((size_t)N * 4);
  float* nrm = (float*)carve((size_t)E * 4);
  float* AGG = (float*)carve((size_t)NH * 4);
  unsigned short* XB  = (unsigned short*)carve((size_t)N * N_F * 2);
  unsigned short* H0  = (unsigned short*)carve((size_t)NH * 2);
  unsigned short* O0  = (unsigned short*)carve((size_t)NH * 2);
  unsigned short* O1  = (unsigned short*)carve((size_t)NH * 2);
  unsigned short* TMP = (unsigned short*)carve((size_t)NH * 2);
  unsigned short* WinT = (unsigned short*)carve((size_t)N_H * N_F * 2);     // [128][512]
  unsigned short* iwT  = (unsigned short*)carve((size_t)L * K * HH * 2);
  unsigned short* dwT  = (unsigned short*)carve((size_t)L * K * HH * 2);
  unsigned short* rwT  = (unsigned short*)carve((size_t)L * T * K * HH * 2);
  unsigned short* WoT  = (unsigned short*)carve((size_t)NPAD_C * N_H * 2);  // [48][128]

  const int TB = 256;
  const int gE  = (E + TB - 1) / TB;
  const int gN  = (N + TB - 1) / TB;
  const int gSp = (int)(((long long)E * 32 + TB - 1) / TB);
  const int gNH = (int)((NH + TB - 1) / TB);
  const int gW  = (N_H * N_H + TB - 1) / TB;

  // ---- one-time conversions (per launch; deterministic) ----
  hipLaunchKernelGGL(f32_to_bf16_kernel, dim3((int)(((long long)N * N_F + TB - 1) / TB)),
                     dim3(TB), 0, stream, x, XB, (long long)N * N_F);
  hipLaunchKernelGGL(transpose_w_kernel, dim3((N_H * N_F + TB - 1) / TB), dim3(TB), 0,
                     stream, Win, WinT, N_F, N_H, N_H);
  for (int i = 0; i < L * K; ++i) {
    hipLaunchKernelGGL(transpose_w_kernel, dim3(gW), dim3(TB), 0, stream,
                       init_w + (size_t)i * HH, iwT + (size_t)i * HH, N_H, N_H, N_H);
    hipLaunchKernelGGL(transpose_w_kernel, dim3(gW), dim3(TB), 0, stream,
                       w_deep + (size_t)i * HH, dwT + (size_t)i * HH, N_H, N_H, N_H);
  }
  for (int i = 0; i < L * T * K; ++i)
    hipLaunchKernelGGL(transpose_w_kernel, dim3(gW), dim3(TB), 0, stream,
                       root_w + (size_t)i * HH, rwT + (size_t)i * HH, N_H, N_H, N_H);
  hipLaunchKernelGGL(transpose_w_kernel, dim3((NPAD_C * N_H + TB - 1) / TB), dim3(TB), 0,
                     stream, Wout, WoT, N_H, N_C, NPAD_C);

  // ---- gcn_norm ----
  hipMemsetAsync(deg, 0, (size_t)N * sizeof(float), stream);
  hipLaunchKernelGGL(deg_kernel,  dim3(gE), dim3(TB), 0, stream, dst, ew, deg, E);
  hipLaunchKernelGGL(dis_kernel,  dim3(gN), dim3(TB), 0, stream, deg, dis, N);
  hipLaunchKernelGGL(norm_kernel, dim3(gE), dim3(TB), 0, stream, src, dst, ew, dis, nrm, E);

  // ---- h = relu(x @ Win + b_in) ----
  launch_gemm<N_F, N_H, true, false, true, false>(XB, WinT, b_in, nullptr, H0, N, stream);

  // ---- L ARMA blocks ----
  for (int l = 0; l < L; ++l) {
    for (int k = 0; k < K; ++k) {
      unsigned short* OUT = (k == 0) ? O0 : O1;

      // out = h @ init_w[l,k]
      launch_gemm<N_H, N_H, false, false, false, false>(
          H0, iwT + (size_t)(l * K + k) * HH, nullptr, nullptr, OUT, N, stream);

      // t = 0: agg = SpMM(out); out = relu(agg + h@root_w[l,0,k] + bias[l,0,k])
      hipMemsetAsync(AGG, 0, (size_t)NH * sizeof(float), stream);
      hipLaunchKernelGGL(spmm_kernel, dim3(gSp), dim3(TB), 0, stream,
                         OUT, src, dst, nrm, AGG, E);
      launch_gemm<N_H, N_H, true, true, true, false>(
          H0, rwT + (size_t)((l * T + 0) * K + k) * HH,
          bias + (size_t)((l * T + 0) * K + k) * N_H, AGG, OUT, N, stream);

      // t = 1: tmp = out @ w[l,0,k]; agg = SpMM(tmp);
      //        out = relu(agg + h@root_w[l,1,k] + bias[l,1,k])
      launch_gemm<N_H, N_H, false, false, false, false>(
          OUT, dwT + (size_t)(l * K + k) * HH, nullptr, nullptr, TMP, N, stream);
      hipMemsetAsync(AGG, 0, (size_t)NH * sizeof(float), stream);
      hipLaunchKernelGGL(spmm_kernel, dim3(gSp), dim3(TB), 0, stream,
                         TMP, src, dst, nrm, AGG, E);
      launch_gemm<N_H, N_H, true, true, true, false>(
          H0, rwT + (size_t)((l * T + 1) * K + k) * HH,
          bias + (size_t)((l * T + 1) * K + k) * N_H, AGG, OUT, N, stream);
    }
    // h = relu(mean over stacks)
    hipLaunchKernelGGL(mean_relu_kernel, dim3(gNH), dim3(TB), 0, stream, O0, O1, H0, NH);
  }

  // ---- logits = h @ Wout + b_out (f32 out) ----
  launch_gemm<N_H, N_C, false, false, true, true>(H0, WoT, b_out, nullptr, out, N, stream);
}
